// HedgehogAttention_20598663152020
// MI455X (gfx1250) — compile-verified, matrix operand order checked by reading, not run on hardware
//
#include <hip/hip_runtime.h>
#include <hip/hip_bf16.h>

#define B 2
#define S 2048
#define D 768
#define H 12
#define HD 64

typedef __attribute__((ext_vector_type(16))) _Float16 v16h;
typedef __attribute__((ext_vector_type(8)))  float    v8f;

// ---- CDNA5 async global->LDS staging (ASYNCcnt path) ----------------------
#if defined(__gfx1250__) && __has_builtin(__builtin_amdgcn_global_load_async_to_lds_b128)
#define USE_ASYNC_LDS 1
#else
#define USE_ASYNC_LDS 0
#endif

#if USE_ASYNC_LDS
typedef int v4i_ __attribute__((vector_size(16)));
__device__ __forceinline__ void async_tile16(void* lds, const void* g) {
  __builtin_amdgcn_global_load_async_to_lds_b128(
      (__attribute__((address_space(1))) v4i_*)g,
      (__attribute__((address_space(3))) v4i_*)lds,
      /*offset=*/0, /*cpol=*/0);
}
__device__ __forceinline__ void wait_async() {
#if __has_builtin(__builtin_amdgcn_s_wait_asynccnt)
  __builtin_amdgcn_s_wait_asynccnt(0);
#else
  asm volatile("s_wait_asynccnt 0" ::: "memory");
#endif
}
#endif

// ---- WMMA helpers ---------------------------------------------------------
// A / B^T loader: row-major [rows x K] tile; lane holds row = row0 + (lane&15),
// halfs [0..7] = K kb..kb+7, [8..15] = K kb+16..kb+23  (kb = (lane>>4)*8)
__device__ __forceinline__ v16h load_frag_row(const _Float16* t, int row0, int col0,
                                              int pitch, int lane) {
  const int m  = lane & 15;
  const int kb = (lane >> 4) << 3;
  const _Float16* p = t + (row0 + m) * pitch + col0 + kb;
  v16h r;
#pragma unroll
  for (int i = 0; i < 8; ++i) { r[i] = p[i]; r[i + 8] = p[i + 16]; }
  return r;
}

// B loader from row-major [K x N] tile (e.g. V): lane holds col = col0 + (lane&15)
__device__ __forceinline__ v16h load_frag_col(const _Float16* t, int k0, int col0,
                                              int pitch, int lane) {
  const int n  = lane & 15;
  const int kb = (lane >> 4) << 3;
  const _Float16* p = t + (k0 + kb) * pitch + col0 + n;
  v16h r;
#pragma unroll
  for (int i = 0; i < 8; ++i) { r[i] = p[i * pitch]; r[i + 8] = p[(i + 16) * pitch]; }
  return r;
}

__device__ __forceinline__ v8f wmma_f16(v16h a, v16h b, v8f c) {
  return __builtin_amdgcn_wmma_f32_16x16x32_f16(false, a, false, b, (short)0, c,
                                                false, false);
}

// ---- 1) QKV projection: q/k/v = h @ W^T + b, stored f16 as [b,h,s,64] ------
__global__ __launch_bounds__(256)
void proj_kernel(const float* __restrict__ hidden,
                 const float* __restrict__ Wq, const float* __restrict__ bq,
                 const float* __restrict__ Wk, const float* __restrict__ bk,
                 const float* __restrict__ Wv, const float* __restrict__ bv,
                 _Float16* __restrict__ q16, _Float16* __restrict__ k16,
                 _Float16* __restrict__ v16) {
  __shared__ _Float16 a_s[64 * 72];
  __shared__ _Float16 b_s[64 * 72];

  const int which = blockIdx.z;
  const float* W    = which == 0 ? Wq : (which == 1 ? Wk : Wv);
  const float* bias = which == 0 ? bq : (which == 1 ? bk : bv);
  _Float16* dst     = which == 0 ? q16 : (which == 1 ? k16 : v16);

  const int tid = threadIdx.x, lane = tid & 31, wave = tid >> 5;
  const int kh = lane >> 4, nl = lane & 15;
  const int mt = blockIdx.x, h = blockIdx.y;
  const int m0 = mt * 64;
  const int mi0 = wave >> 2, ni0 = wave & 3;
  const int mi1 = mi0 + 2,   ni1 = ni0;

  v8f acc0 = {}, acc1 = {};

  for (int k0 = 0; k0 < D; k0 += 64) {
    for (int i = tid; i < 64 * 64; i += 256) {
      int r = i >> 6, c = i & 63;
      a_s[r * 72 + c] = (_Float16)hidden[(size_t)(m0 + r) * D + k0 + c];
      b_s[r * 72 + c] = (_Float16)W[(size_t)(h * 64 + r) * D + k0 + c];
    }
    __syncthreads();
#pragma unroll
    for (int kc = 0; kc < 2; ++kc) {
      acc0 = wmma_f16(load_frag_row(a_s, mi0 * 16, kc * 32, 72, lane),
                      load_frag_row(b_s, ni0 * 16, kc * 32, 72, lane), acc0);
      acc1 = wmma_f16(load_frag_row(a_s, mi1 * 16, kc * 32, 72, lane),
                      load_frag_row(b_s, ni1 * 16, kc * 32, 72, lane), acc1);
    }
    __syncthreads();
  }

  const int mis[2] = {mi0, mi1}, nis[2] = {ni0, ni1};
  v8f ac[2] = {acc0, acc1};
#pragma unroll
  for (int t = 0; t < 2; ++t) {
#pragma unroll
    for (int r = 0; r < 8; ++r) {
      int m   = m0 + mis[t] * 16 + kh * 8 + r;
      int col = nis[t] * 16 + nl;
      int b   = m >> 11, s = m & (S - 1);
      float v = ac[t][r] + bias[h * 64 + col];
      dst[(((size_t)b * H + h) * S + s) * 64 + col] = (_Float16)v;
    }
  }
}

// ---- 2) hedgehog feature map: f = [exp(y), exp(-y)], y = x @ Wf^T + bf ----
__global__ __launch_bounds__(256)
void feature_kernel(const _Float16* __restrict__ src, const float* __restrict__ Wf,
                    const float* __restrict__ bf, _Float16* __restrict__ dst) {
  __shared__ float    w_s[64 * 64];
  __shared__ float    b_s[64];
  __shared__ _Float16 x_s[64 * 64];

  const int tid = threadIdx.x;
  const int bh = blockIdx.y, st = blockIdx.x;
  const size_t row0 = (size_t)bh * S + st * 64;
  const _Float16* sp = src + row0 * 64;

#if USE_ASYNC_LDS
  for (int c = tid; c < 512; c += 256)
    async_tile16((char*)x_s + c * 16, (const char*)sp + c * 16);
#else
  for (int i = tid; i < 4096; i += 256) x_s[i] = sp[i];
#endif
  for (int i = tid; i < 4096; i += 256) w_s[i] = Wf[i];
  if (tid < 64) b_s[tid] = bf[tid];
#if USE_ASYNC_LDS
  wait_async();
#endif
  __syncthreads();

  for (int i = 0; i < 16; ++i) {
    int idx = i * 256 + tid;
    int r = idx >> 6, d = idx & 63;
    float y = b_s[d];
#pragma unroll 8
    for (int e = 0; e < 64; ++e) y += (float)x_s[r * 64 + e] * w_s[d * 64 + e];
    size_t off = (row0 + r) * 128;
    dst[off + d]      = (_Float16)__expf(y);
    dst[off + 64 + d] = (_Float16)__expf(-y);
  }
}

// ---- 3) per-(b,h) column sums of fk: denominator trick for pred_attns -----
__global__ void sumfk_kernel(const _Float16* __restrict__ fk,
                             float* __restrict__ sumfk) {
  int bh = blockIdx.x, d = threadIdx.x;
  const _Float16* p = fk + (size_t)bh * S * 128 + d;
  float s = 0.f;
  for (int n = 0; n < S; ++n) s += (float)p[(size_t)n * 128];
  sumfk[bh * 128 + d] = s;
}

// ---- 4) two-pass tiled attention ------------------------------------------
__global__ __launch_bounds__(256)
void attn_kernel(const _Float16* __restrict__ q16, const _Float16* __restrict__ k16,
                 const _Float16* __restrict__ v16, const _Float16* __restrict__ fq16,
                 const _Float16* __restrict__ fk16, const float* __restrict__ sumfk,
                 float* __restrict__ outp, float* __restrict__ predp,
                 float* __restrict__ truep) {
  __shared__ __align__(16) char smem[9216 + 9216 + 34816];
  __shared__ float rowsum[64];
  __shared__ float part[256];
  __shared__ float denom[64];
  __shared__ float sfk[128];

  _Float16* q_s = (_Float16*)smem;
  _Float16* k_s = (_Float16*)(smem + 9216);
  char* un = smem + 18432;
  _Float16* v_s  = (_Float16*)un;            // pass 1
  _Float16* e_s  = (_Float16*)(un + 9216);   // pass 1
  _Float16* fq_s = (_Float16*)un;            // pass 2
  _Float16* fk_s = (_Float16*)(un + 17408);  // pass 2

  const int tid = threadIdx.x, lane = tid & 31, wave = tid >> 5;
  const int bh = blockIdx.y, mt = blockIdx.x, m0 = mt * 64;
  const int bb = bh / H, hh = bh % H;
  const int kh = lane >> 4, nl = lane & 15;
  const int mi0 = wave >> 2, ni0 = wave & 3;
  const int mi1 = mi0 + 2,   ni1 = ni0;

  const _Float16* qp  = q16  + ((size_t)bh * S + m0) * 64;
  const _Float16* fqp = fq16 + ((size_t)bh * S + m0) * 128;
  const _Float16* kb_ = k16  + (size_t)bh * S * 64;
  const _Float16* vb_ = v16  + (size_t)bh * S * 64;
  const _Float16* fkb = fk16 + (size_t)bh * S * 128;

#if USE_ASYNC_LDS
  for (int c = tid; c < 512; c += 256)
    async_tile16((char*)q_s + (c >> 3) * 144 + (c & 7) * 16,
                 (const char*)qp + c * 16);
#else
  for (int i = tid; i < 64 * 64; i += 256) q_s[(i >> 6) * 72 + (i & 63)] = qp[i];
#endif
  if (tid < 128) sfk[tid] = sumfk[bh * 128 + tid];
  if (tid < 64) rowsum[tid] = 0.f;
#if USE_ASYNC_LDS
  wait_async();
#endif
  __syncthreads();

  v8f oacc0 = {}, oacc1 = {};

  // pass 1: rowsum(exp(QK^T/8)) and unnormalized exp(QK^T/8) @ V
  for (int n0 = 0; n0 < S; n0 += 64) {
    const _Float16* kp = kb_ + (size_t)n0 * 64;
    const _Float16* vp = vb_ + (size_t)n0 * 64;
#if USE_ASYNC_LDS
    for (int c = tid; c < 512; c += 256) {
      int lo = (c >> 3) * 144 + (c & 7) * 16;
      async_tile16((char*)k_s + lo, (const char*)kp + c * 16);
      async_tile16((char*)v_s + lo, (const char*)vp + c * 16);
    }
    wait_async();
#else
    for (int i = tid; i < 64 * 64; i += 256) {
      int r = i >> 6, c = i & 63;
      k_s[r * 72 + c] = kp[i];
      v_s[r * 72 + c] = vp[i];
    }
#endif
    if (n0 + 64 < S) {
      __builtin_prefetch(kp + 4096 + (tid << 4), 0, 1);
      __builtin_prefetch(vp + 4096 + (tid << 4), 0, 1);
    }
    __syncthreads();

    {
      v8f c0 = {}, c1 = {};
      c0 = wmma_f16(load_frag_row(q_s, mi0 * 16, 0, 72, lane),
                    load_frag_row(k_s, ni0 * 16, 0, 72, lane), c0);
      c0 = wmma_f16(load_frag_row(q_s, mi0 * 16, 32, 72, lane),
                    load_frag_row(k_s, ni0 * 16, 32, 72, lane), c0);
      c1 = wmma_f16(load_frag_row(q_s, mi1 * 16, 0, 72, lane),
                    load_frag_row(k_s, ni1 * 16, 0, 72, lane), c1);
      c1 = wmma_f16(load_frag_row(q_s, mi1 * 16, 32, 72, lane),
                    load_frag_row(k_s, ni1 * 16, 32, 72, lane), c1);
#pragma unroll
      for (int r = 0; r < 8; ++r) {
        e_s[(mi0 * 16 + kh * 8 + r) * 72 + ni0 * 16 + nl] =
            (_Float16)__expf(c0[r] * 0.125f);
        e_s[(mi1 * 16 + kh * 8 + r) * 72 + ni1 * 16 + nl] =
            (_Float16)__expf(c1[r] * 0.125f);
      }
    }
    __syncthreads();

    {
      int row = tid & 63, ch = tid >> 6;
      float p = 0.f;
#pragma unroll
      for (int c = 0; c < 16; ++c) p += (float)e_s[row * 72 + ch * 16 + c];
      part[row * 4 + ch] = p;
    }
    __syncthreads();
    if (tid < 64)
      rowsum[tid] += part[tid * 4] + part[tid * 4 + 1] + part[tid * 4 + 2] +
                     part[tid * 4 + 3];

#pragma unroll
    for (int kc = 0; kc < 2; ++kc) {
      oacc0 = wmma_f16(load_frag_row(e_s, mi0 * 16, kc * 32, 72, lane),
                       load_frag_col(v_s, kc * 32, ni0 * 16, 72, lane), oacc0);
      oacc1 = wmma_f16(load_frag_row(e_s, mi1 * 16, kc * 32, 72, lane),
                       load_frag_col(v_s, kc * 32, ni1 * 16, 72, lane), oacc1);
    }
    __syncthreads();
  }

  // load fq tile (reuses pass-1 LDS), compute pred denominators = fq . sum_fk
#if USE_ASYNC_LDS
  for (int c = tid; c < 1024; c += 256)
    async_tile16((char*)fq_s + (c >> 4) * 272 + (c & 15) * 16,
                 (const char*)fqp + c * 16);
  wait_async();
#else
  for (int i = tid; i < 64 * 128; i += 256)
    fq_s[(i >> 7) * 136 + (i & 127)] = fqp[i];
#endif
  __syncthreads();
  if (tid < 64) {
    float d = 0.f;
    for (int e = 0; e < 128; ++e) d += (float)fq_s[tid * 136 + e] * sfk[e];
    denom[tid] = d;
  }
  __syncthreads();

  // pass 2: recompute scores, stream out normalized true & pred attention
  const int mis[2] = {mi0, mi1}, nis[2] = {ni0, ni1};
  for (int n0 = 0; n0 < S; n0 += 64) {
    const _Float16* kp  = kb_ + (size_t)n0 * 64;
    const _Float16* fkp = fkb + (size_t)n0 * 128;
#if USE_ASYNC_LDS
    for (int c = tid; c < 512; c += 256)
      async_tile16((char*)k_s + (c >> 3) * 144 + (c & 7) * 16,
                   (const char*)kp + c * 16);
    for (int c = tid; c < 1024; c += 256)
      async_tile16((char*)fk_s + (c >> 4) * 272 + (c & 15) * 16,
                   (const char*)fkp + c * 16);
    wait_async();
#else
    for (int i = tid; i < 64 * 64; i += 256)
      k_s[(i >> 6) * 72 + (i & 63)] = kp[i];
    for (int i = tid; i < 64 * 128; i += 256)
      fk_s[(i >> 7) * 136 + (i & 127)] = fkp[i];
#endif
    if (n0 + 64 < S) __builtin_prefetch(fkp + 8192 + (tid << 5), 0, 1);
    __syncthreads();

#pragma unroll
    for (int t = 0; t < 2; ++t) {
      int mi = mis[t], ni = nis[t];
      v8f c = {};
      c = wmma_f16(load_frag_row(q_s, mi * 16, 0, 72, lane),
                   load_frag_row(k_s, ni * 16, 0, 72, lane), c);
      c = wmma_f16(load_frag_row(q_s, mi * 16, 32, 72, lane),
                   load_frag_row(k_s, ni * 16, 32, 72, lane), c);
      v8f p = {};
#pragma unroll
      for (int kc = 0; kc < 4; ++kc)
        p = wmma_f16(load_frag_row(fq_s, mi * 16, kc * 32, 136, lane),
                     load_frag_row(fk_s, ni * 16, kc * 32, 136, lane), p);
#pragma unroll
      for (int r = 0; r < 8; ++r) {
        int ml = mi * 16 + kh * 8 + r;
        size_t row = (size_t)bh * S + (m0 + ml);
        int ng = n0 + ni * 16 + nl;
        truep[row * S + ng] = __expf(c[r] * 0.125f) / rowsum[ml];
        predp[row * S + ng] = p[r] / denom[ml];
      }
    }
    __syncthreads();
  }

  // outputs = (exp(QK^T/8) @ V) / rowsum, interleaved back to [B,S,D]
  {
    v8f oa[2] = {oacc0, oacc1};
#pragma unroll
    for (int t = 0; t < 2; ++t) {
#pragma unroll
      for (int r = 0; r < 8; ++r) {
        int ml = mis[t] * 16 + kh * 8 + r;
        int dcol = hh * 64 + nis[t] * 16 + nl;
        outp[((size_t)bb * S + (m0 + ml)) * D + dcol] = oa[t][r] / rowsum[ml];
      }
    }
  }
}

extern "C" void kernel_launch(void* const* d_in, const int* in_sizes, int n_in,
                              void* d_out, int out_size, void* d_ws, size_t ws_size,
                              hipStream_t stream) {
  (void)in_sizes; (void)n_in; (void)out_size; (void)ws_size;
  const float* hidden = (const float*)d_in[0];
  const float* Wq  = (const float*)d_in[1];
  const float* bq  = (const float*)d_in[2];
  const float* Wk  = (const float*)d_in[3];
  const float* bk  = (const float*)d_in[4];
  const float* Wv  = (const float*)d_in[5];
  const float* bv  = (const float*)d_in[6];
  const float* Wfq = (const float*)d_in[7];
  const float* bfq = (const float*)d_in[8];
  const float* Wfk = (const float*)d_in[9];
  const float* bfk = (const float*)d_in[10];

  const size_t QKV = (size_t)B * H * S * HD;       // 3,145,728 halfs each
  const size_t FEA = (size_t)B * H * S * 2 * HD;   // 6,291,456 halfs each
  _Float16* q16  = (_Float16*)d_ws;
  _Float16* k16  = q16 + QKV;
  _Float16* v16  = k16 + QKV;
  _Float16* fq16 = v16 + QKV;
  _Float16* fk16 = fq16 + FEA;
  float*    sfk  = (float*)(fk16 + FEA);           // B*H*128 floats

  float* outp  = (float*)d_out;
  float* predp = outp + (size_t)B * S * D;
  float* truep = predp + (size_t)B * H * S * S;

  proj_kernel<<<dim3(B * S / 64, H, 3), 256, 0, stream>>>(
      hidden, Wq, bq, Wk, bk, Wv, bv, q16, k16, v16);
  feature_kernel<<<dim3(S / 64, B * H), 256, 0, stream>>>(q16, Wfq, bfq, fq16);
  feature_kernel<<<dim3(S / 64, B * H), 256, 0, stream>>>(k16, Wfk, bfk, fk16);
  sumfk_kernel<<<B * H, 128, 0, stream>>>(fk16, sfk);
  attn_kernel<<<dim3(S / 64, B * H), 256, 0, stream>>>(
      q16, k16, v16, fq16, fk16, sfk, outp, predp, truep);
}
